// RecursiveMultiHorizonHead_56573309223342
// MI455X (gfx1250) — matrix operand order repaired and checked by reading, not compiled, not run on hardware
//
#include <hip/hip_runtime.h>

// Fixed problem sizes from the reference
#define BATCH 4096
#define HDIM  256
#define GDIM  1024          // 4*H
#define GROWS 1040          // GDIM + one 16-row tile holding proj_W (pred column)
#define TSTEP 96
#define BM    32            // batch rows per workgroup (rows are independent)
#define HPAD  264           // padded LDS row length in halfs for h (528B, 16B aligned)
#define WLP   136           // padded LDS row length in halfs for W cache (272B)
#define KLDS  4             // k-tiles (of 32) served from LDS; rest stream from L2
// Dynamic LDS: W cache (K<128 half) + double-buffered h
#define SMEM_HALFS (GROWS * WLP + 2 * BM * HPAD)
#define SMEM_BYTES (SMEM_HALFS * 2)            // 282880 + 33792 = 316672 <= 320KB

typedef __attribute__((ext_vector_type(16))) _Float16 v16h;
typedef __attribute__((ext_vector_type(8)))  _Float16 v8h;
typedef __attribute__((ext_vector_type(8)))  float    v8f;

static __device__ __forceinline__ v16h cat8(v8h a, v8h b) {
  return __builtin_shufflevector(a, b, 0,1,2,3,4,5,6,7,8,9,10,11,12,13,14,15);
}
static __device__ __forceinline__ float fsig(float x)  { return 1.0f / (1.0f + __expf(-x)); }
static __device__ __forceinline__ float ftanh(float x) { return 2.0f / (1.0f + __expf(-2.0f * x)) - 1.0f; }
static __device__ __forceinline__ v8f vzero() {
  v8f z;
#pragma unroll
  for (int i = 0; i < 8; ++i) z[i] = 0.0f;
  return z;
}

// ---------------------------------------------------------------------------
// Prep kernel 1: build W_eff (f16, gate-interleaved rows) and bias_eff (f32).
// New row R = (4*hcTile + gate)*16 + j maps original gate row
// (gate*256 + hcTile*16 + j). Rows 1024..1039: row 1024 = proj_W, rest zero
// (pred = h @ proj_W.T appears as column 0 of output tile 64).
// ---------------------------------------------------------------------------
__global__ void build_weff_kernel(const float* __restrict__ W_ih,
                                  const float* __restrict__ W_hh,
                                  const float* __restrict__ b_ih,
                                  const float* __restrict__ b_hh,
                                  const float* __restrict__ proj_W,
                                  const float* __restrict__ proj_b,
                                  _Float16* __restrict__ Weff,
                                  float* __restrict__ biasE)
{
  const int R = blockIdx.x;     // 0..1039
  const int k = threadIdx.x;    // 0..255
  if (R < GDIM) {
    const int ntile = R >> 4, j = R & 15;
    const int hc = ntile >> 2, gate = ntile & 3;
    const int orig = gate * HDIM + hc * 16 + j;
    // fold rank-1 input path: xin_t = h_{t-1}@proj_W.T + proj_b
    const float wv = W_hh[(size_t)orig * HDIM + k] + W_ih[orig] * proj_W[k];
    Weff[(size_t)R * HDIM + k] = (_Float16)wv;
    if (k == 0)
      biasE[R] = b_ih[orig] + b_hh[orig] + proj_b[0] * W_ih[orig];
  } else {
    const int j = R - GDIM;
    Weff[(size_t)R * HDIM + k] = (j == 0) ? (_Float16)proj_W[k] : (_Float16)0.0f;
  }
}

// ---------------------------------------------------------------------------
// Prep kernel 2: closed-form step 0 (h=c=xin=0 -> gates = b_ih+b_hh, same for
// every batch row). Produces h0 (f16) and c0 (f32), each [H].
// ---------------------------------------------------------------------------
__global__ void build_state0_kernel(const float* __restrict__ b_ih,
                                    const float* __restrict__ b_hh,
                                    float* __restrict__ c0,
                                    _Float16* __restrict__ h0)
{
  const int j = threadIdx.x;
  const float bi = b_ih[j]           + b_hh[j];
  const float bg = b_ih[2*HDIM + j]  + b_hh[2*HDIM + j];
  const float bo = b_ih[3*HDIM + j]  + b_hh[3*HDIM + j];
  const float c  = fsig(bi) * ftanh(bg);
  c0[j] = c;
  h0[j] = (_Float16)(fsig(bo) * ftanh(c));
}

// ---------------------------------------------------------------------------
// One h-column group (16 cols x 4 gates = 4 contiguous 16x16 N-tiles) for both
// 16-row tiles of this workgroup. B-fragments: kt<KLDS from LDS W-cache,
// kt>=KLDS streamed from L2. FULL = gates + state update; PRED = pred column.
// ---------------------------------------------------------------------------
template<bool FULL, bool PRED>
static __device__ __forceinline__ void do_hg(
    int hc, const _Float16* __restrict__ Weff, const _Float16* __restrict__ wlds,
    const _Float16* __restrict__ hread, _Float16* __restrict__ hwrite,
    float b0, float b1, float b2, float b3,
    v8f& cr0, v8f& cr1,
    int lane, int ci, int rbse, float pb,
    float* __restrict__ out, int wgb, int t)
{
  v8f acc[4][2];
  v8f accp[2];
#pragma unroll
  for (int rt = 0; rt < 2; ++rt) {
    accp[rt] = vzero();
#pragma unroll
    for (int g = 0; g < 4; ++g) acc[g][rt] = vzero();
  }
  const int koffB = (lane & 16) ? 16 : 0;   // B-fragment K offset (halfs)
  const int koffA = (lane & 16) ? 16 : 0;   // A-fragment K offset (halfs)

#pragma unroll
  for (int kt = 0; kt < 8; ++kt) {          // K = 256 = 8 x 32
    v16h Bf[4];
    if constexpr (FULL) {
#pragma unroll
      for (int g = 0; g < 4; ++g) {
        // B[k,n] = W_eff[row=ntile*16+n][k]: lane n holds 32 contiguous bytes
        const int rowW = (4 * hc + g) * 16 + ci;
        if (kt < KLDS) {                    // compile-time resolved (unrolled)
          const v8h* p = (const v8h*)(wlds + rowW * WLP + kt * 32 + koffB);
          Bf[g] = cat8(p[0], p[1]);
        } else {
          const v8h* p = (const v8h*)(Weff + (size_t)rowW * HDIM + kt * 32 + koffB);
          Bf[g] = cat8(p[0], p[1]);
        }
      }
    }
    v16h Bp;
    if constexpr (PRED) {
      const int rowW = GDIM + ci;           // proj_W row tile
      if (kt < KLDS) {
        const v8h* p = (const v8h*)(wlds + rowW * WLP + kt * 32 + koffB);
        Bp = cat8(p[0], p[1]);
      } else {
        const v8h* p = (const v8h*)(Weff + (size_t)rowW * HDIM + kt * 32 + koffB);
        Bp = cat8(p[0], p[1]);
      }
    }
#pragma unroll
    for (int rt = 0; rt < 2; ++rt) {
      // A-fragment: lane<16 rows M=lane K {0..15}; lane>=16 K {16..31}
      const _Float16* ha = hread + (rt * 16 + ci) * HPAD + kt * 32 + koffA;
      const v16h Af = cat8(*(const v8h*)ha, *(const v8h*)(ha + 8));
      if constexpr (FULL) {
#pragma unroll
        for (int g = 0; g < 4; ++g)
          acc[g][rt] = __builtin_amdgcn_wmma_f32_16x16x32_f16(
              false, Af, false, Bf[g], (short)0, acc[g][rt], false, false);
      }
      if constexpr (PRED)
        accp[rt] = __builtin_amdgcn_wmma_f32_16x16x32_f16(
            false, Af, false, Bp, (short)0, accp[rt], false, false);
    }
  }

  if constexpr (FULL) {
    // Elementwise LSTM cell; c stays in registers in accumulator layout.
#pragma unroll
    for (int rt = 0; rt < 2; ++rt) {
      v8f& cr = rt ? cr1 : cr0;
#pragma unroll
      for (int r = 0; r < 8; ++r) {
        const float xi = acc[0][rt][r] + b0;
        const float xf = acc[1][rt][r] + b1;
        const float xg = acc[2][rt][r] + b2;
        const float xo = acc[3][rt][r] + b3;
        const float cn = fsig(xf) * cr[r] + fsig(xi) * ftanh(xg);
        cr[r] = cn;
        const float hn = fsig(xo) * ftanh(cn);
        hwrite[(rt * 16 + rbse + r) * HPAD + hc * 16 + ci] = (_Float16)hn;
      }
    }
  }
  if constexpr (PRED) {
    if (ci == 0) {                           // lanes 0 and 16 hold column 0
#pragma unroll
      for (int rt = 0; rt < 2; ++rt)
#pragma unroll
        for (int r = 0; r < 8; ++r)
          out[(size_t)(wgb + rt * 16 + rbse + r) * TSTEP + (t - 1)] = accp[rt][r] + pb;
    }
  }
}

// ---------------------------------------------------------------------------
// Main kernel: 128 WGs x 256 threads (8 wave32). Wave w owns h-columns
// [32w, 32w+32). 96 fully in-WGP recurrent steps, one barrier per step.
// Uses 316KB of CDNA5's 320KB LDS: half of W_eff cached + h ping-pong.
// ---------------------------------------------------------------------------
__global__ void __launch_bounds__(256)
lstm_decode_kernel(const _Float16* __restrict__ Weff,
                   const float* __restrict__ biasE,
                   const float* __restrict__ c0,
                   const _Float16* __restrict__ h0,
                   const float* __restrict__ projb,
                   float* __restrict__ out)
{
  extern __shared__ _Float16 smem[];
  _Float16* wlds = smem;                       // [GROWS][WLP]   K in [0,128)
  _Float16* hbuf = smem + GROWS * WLP;         // [2][BM][HPAD]  ping-pong h

  const int tid  = threadIdx.x;
  const int lane = tid & 31;
  const int w    = tid >> 5;
  const int ci   = lane & 15;
  const int rbse = (lane & 16) ? 8 : 0;
  const int wgb  = blockIdx.x * BM;

  // Warm L2 with W_eff (it stays L2-resident for the whole kernel).
  if (blockIdx.x < 4) {
    const char* p = (const char*)Weff;
    for (int off = tid * 512; off < GROWS * HDIM * 2; off += 256 * 512)
      __builtin_prefetch(p + off, 0, 1);
  }

  // Cache the K<64 quarter of W_eff in LDS (133KB of payload, padded rows).
  for (int idx = tid; idx < GROWS * 8; idx += 256) {
    const int row = idx >> 3, ch = idx & 7;              // 8 x 16B chunks/row
    const v8h v = *(const v8h*)(Weff + (size_t)row * HDIM + ch * 8);
    *(v8h*)(wlds + row * WLP + ch * 8) = v;
  }

  // h_0 is identical across batch rows.
  for (int idx = tid; idx < BM * HDIM; idx += 256) {
    const int row = idx >> 8, col = idx & (HDIM - 1);
    hbuf[row * HPAD + col] = h0[col];
  }

  float bias_r[2][4];
  v8f creg[2][2];
#pragma unroll
  for (int hg = 0; hg < 2; ++hg) {
    const int hc = 2 * w + hg;
#pragma unroll
    for (int g = 0; g < 4; ++g) bias_r[hg][g] = biasE[(4 * hc + g) * 16 + ci];
    const float cv = c0[hc * 16 + ci];
    v8f cs;
#pragma unroll
    for (int i = 0; i < 8; ++i) cs[i] = cv;
    creg[hg][0] = cs;
    creg[hg][1] = cs;
  }
  const float pb = projb[0];
  __syncthreads();

  for (int t = 1; t <= TSTEP; ++t) {
    const _Float16* hread = hbuf + ((t + 1) & 1) * (BM * HPAD);
    _Float16*      hwrite = hbuf + (t & 1) * (BM * HPAD);
    const bool full = (t < TSTEP);
    if (full) {
      if (w == 7) {
        do_hg<true, true >(14, Weff, wlds, hread, hwrite,
                           bias_r[0][0], bias_r[0][1], bias_r[0][2], bias_r[0][3],
                           creg[0][0], creg[0][1], lane, ci, rbse, pb, out, wgb, t);
        do_hg<true, false>(15, Weff, wlds, hread, hwrite,
                           bias_r[1][0], bias_r[1][1], bias_r[1][2], bias_r[1][3],
                           creg[1][0], creg[1][1], lane, ci, rbse, pb, out, wgb, t);
      } else {
        do_hg<true, false>(2 * w, Weff, wlds, hread, hwrite,
                           bias_r[0][0], bias_r[0][1], bias_r[0][2], bias_r[0][3],
                           creg[0][0], creg[0][1], lane, ci, rbse, pb, out, wgb, t);
        do_hg<true, false>(2 * w + 1, Weff, wlds, hread, hwrite,
                           bias_r[1][0], bias_r[1][1], bias_r[1][2], bias_r[1][3],
                           creg[1][0], creg[1][1], lane, ci, rbse, pb, out, wgb, t);
      }
    } else if (w == 7) {
      // Tail: last pred_{T-1} = h_{T-1} @ proj_W.T + proj_b (pred tile only).
      do_hg<false, true>(14, Weff, wlds, hread, hwrite,
                         0.f, 0.f, 0.f, 0.f,
                         creg[0][0], creg[0][1], lane, ci, rbse, pb, out, wgb, t);
    }
    __syncthreads();
  }
}

// ---------------------------------------------------------------------------
// Workspace layout (bytes):
//   [0)            W_eff  f16  1040*256*2 = 532480
//   [532480)       biasE  f32  1024*4     =   4096
//   [536576)       c0     f32   256*4     =   1024
//   [537600)       h0     f16   256*2     =    512
// ---------------------------------------------------------------------------
extern "C" void kernel_launch(void* const* d_in, const int* in_sizes, int n_in,
                              void* d_out, int out_size, void* d_ws, size_t ws_size,
                              hipStream_t stream) {
  (void)in_sizes; (void)n_in; (void)out_size; (void)ws_size;
  const float* W_ih  = (const float*)d_in[3];
  const float* W_hh  = (const float*)d_in[4];
  const float* b_ih  = (const float*)d_in[5];
  const float* b_hh  = (const float*)d_in[6];
  const float* projW = (const float*)d_in[7];
  const float* projb = (const float*)d_in[8];
  // d_in[0..2] (x, enc_W, enc_b) feed dead code in the reference -> skipped.

  char* ws = (char*)d_ws;
  _Float16* Weff  = (_Float16*)(ws);
  float*    biasE = (float*)(ws + 532480);
  float*    c0    = (float*)(ws + 536576);
  _Float16* h0    = (_Float16*)(ws + 537600);

  build_weff_kernel<<<GROWS, 256, 0, stream>>>(W_ih, W_hh, b_ih, b_hh, projW, projb, Weff, biasE);
  build_state0_kernel<<<1, 256, 0, stream>>>(b_ih, b_hh, c0, h0);
  lstm_decode_kernel<<<BATCH / BM, 256, SMEM_BYTES, stream>>>(Weff, biasE, c0, h0, projb,
                                                              (float*)d_out);
}
